// InterpretableMultiHeadAttention_31739808318081
// MI455X (gfx1250) — matrix operand order repaired and checked by reading, not compiled, not run on hardware
//
#include <hip/hip_runtime.h>
#include <hip/hip_bf16.h>
#include <stdint.h>

// Problem constants (match reference)
#define B  2
#define S  2048
#define D  1024
#define H  16
#define DH 64

typedef _Float16 half_t;
typedef __attribute__((ext_vector_type(16))) _Float16 v16h;
typedef __attribute__((ext_vector_type(8)))  float    v8f;

union HFrag { v16h v; uint32_t u[8]; };

// ---------------------------------------------------------------------------
// WMMA wrapper: D = A(16x32 f16) * B(32x16 f16) + C(16x16 f32)
// ---------------------------------------------------------------------------
__device__ inline v8f wmma_f16(v16h a, v16h b, v8f c) {
  return __builtin_amdgcn_wmma_f32_16x16x32_f16(
      /*neg_a=*/false, a, /*neg_b=*/false, b,
      /*c_mod=*/(short)0, c, /*reuse_a=*/false, /*reuse_b=*/false);
}

// ---------------------------------------------------------------------------
// A-fragment loader (16x32, row-major source, rowStride in halves).
// Lane L<16 : row L,   K = {0..15}
// Lane L>=16: row L-16, K = {16..31}
// (contiguous order, matching load_b_frag)
// ---------------------------------------------------------------------------
__device__ inline v16h load_a_frag(const half_t* base, int rowStride, int kbase) {
  int lane = threadIdx.x & 31;
  int row  = lane & 15;
  int koff = (lane >> 4) << 4;   // 0 or 16
  const half_t* rp = base + (size_t)row * rowStride + kbase;
  HFrag f;
#pragma unroll
  for (int r = 0; r < 8; ++r) {
    int k = koff + 2 * r;
    f.u[r] = *(const uint32_t*)(rp + k);
  }
  return f.v;
}

// ---------------------------------------------------------------------------
// B-fragment loader (32x16). Lane n (n=lane&15) holds column n with
// K = (lane>>4)*16 + {0..15}, contiguous halves at colPtr (caller bakes in
// the per-lane column offset).
// ---------------------------------------------------------------------------
__device__ inline v16h load_b_frag(const half_t* colPtr) {
  int lane = threadIdx.x & 31;
  int kf = (lane >> 4) << 4;     // 0 or 16
  HFrag f;
#pragma unroll
  for (int r = 0; r < 8; ++r)
    f.u[r] = *(const uint32_t*)(colPtr + kf + 2 * r);
  return f.v;
}

// ---------------------------------------------------------------------------
// Tensor Data Mover: DMA a 2D f16 tile [rows x 64] (row stride 64 halves in
// memory) into LDS, padded to 72-half rows (pad_interval=32 dw, pad_amount=
// 4 dw) so per-column LDS reads are bank-conflict free.
// This toolchain carries the 6-arg builtin (clang-23 / therock-10.0 form):
//   (uint32x4 g0, int32x8 g1, int32x4 g2, int32x4 g3, int32x8 extra, i32 cpol)
// ---------------------------------------------------------------------------
#if defined(__has_builtin)
#if __has_builtin(__builtin_amdgcn_tensor_load_to_lds) && \
    __has_builtin(__builtin_amdgcn_s_wait_tensorcnt)
#define HAVE_TDM 1
#endif
#endif

#ifdef HAVE_TDM
typedef uint32_t u32x4 __attribute__((ext_vector_type(4)));
typedef int      i32x4 __attribute__((ext_vector_type(4)));
typedef int      i32x8 __attribute__((ext_vector_type(8)));

__device__ inline void tdm_load_k_tile(uint32_t lds_byte_addr, const void* gaddr,
                                       uint32_t tile_rows) {
  uint64_t ga = (uint64_t)(uintptr_t)gaddr;
  u32x4 g0;
  g0[0] = 1u;                                    // count=1 (valid user D#)
  g0[1] = lds_byte_addr;                         // lds_addr (bits 63:32)
  g0[2] = (uint32_t)(ga & 0xffffffffu);          // global_addr[31:0]
  g0[3] = (uint32_t)((ga >> 32) & 0x01ffffffu)   // global_addr[56:32]
        | (2u << 30);                            // type=2 ("image")
  i32x8 g1;
  g1[0] = (1 << 16)                              // data_size = 2 bytes
        | (1 << 20)                              // pad_enable
        | (4 << 22)                              // pad_interval: 32 dwords
        | (3 << 25);                             // pad_amount : 4 dwords
  g1[1] = (int)(64u << 16);                      // tensor_dim0[15:0]=64 (bits 63:48)
  g1[2] = (int)((uint32_t)S << 16);              // dim0[31:16]=0 | tensor_dim1[15:0]=S
  g1[3] = (int)(64u << 16);                      // dim1[31:16]=0 | tile_dim0=64
  g1[4] = (int)(tile_rows & 0xffffu);            // tile_dim1=rows; tile_dim2=0
  g1[5] = 64;                                    // tensor_dim0_stride[31:0]=64
  g1[6] = 0;                                     // stride0[47:32] | stride1[15:0]
  g1[7] = 0;                                     // stride1[47:16]
  i32x4 z4 = (i32x4){0, 0, 0, 0};
  i32x8 z8 = (i32x8){0, 0, 0, 0, 0, 0, 0, 0};
  __builtin_amdgcn_tensor_load_to_lds(g0, g1, z4, z4, z8, 0);
}
#endif

// ---------------------------------------------------------------------------
// Kernel 1: projection GEMM.  out[b,h,s,e] = sum_d X[b,s,d] * W[h,e,d] + bias
// 64 s-rows x 64 e-cols per workgroup, 4 waves (one 16-row strip each).
// TRANS: write out[b,e,s] (V projection -> contiguous columns for P*V).
// ---------------------------------------------------------------------------
template <int HN, bool TRANS>
__global__ __launch_bounds__(128) void proj_kernel(
    const float* __restrict__ X, const float* __restrict__ W,
    const float* __restrict__ bias, half_t* __restrict__ out)
{
  __shared__ half_t As[64][32];
  __shared__ half_t Bs[64][32];
  int tid  = threadIdx.x;
  int wave = tid >> 5, lane = tid & 31;
  int s0 = blockIdx.x * 64;
  int bh = blockIdx.y;
  int b = bh / HN, h = bh % HN;
  const float* Xb = X + ((size_t)b * S + s0) * D;
  const float* Wh = W + (size_t)h * DH * D;

  v8f acc[4];
#pragma unroll
  for (int t = 0; t < 4; ++t) acc[t] = (v8f){0,0,0,0,0,0,0,0};

  for (int d0 = 0; d0 < D; d0 += 32) {
    if (d0 + 32 < D) {             // cover latency of the next f32 stripe
      int r = tid >> 2, c = (tid & 3) << 3;
      __builtin_prefetch(&Xb[(size_t)r * D + d0 + 32 + c], 0, 0);
      __builtin_prefetch(&Wh[(size_t)r * D + d0 + 32 + c], 0, 0);
    }
#pragma unroll
    for (int i = 0; i < 16; ++i) {            // 2048 elems / 128 threads
      int idx = tid + i * 128;
      int r = idx >> 5, c = idx & 31;
      As[r][c] = (half_t)Xb[(size_t)r * D + d0 + c];
      Bs[r][c] = (half_t)Wh[(size_t)r * D + d0 + c];
    }
    __syncthreads();
    v16h a = load_a_frag(&As[wave * 16][0], 32, 0);
#pragma unroll
    for (int t = 0; t < 4; ++t) {
      v16h bb = load_b_frag(&Bs[t * 16 + (lane & 15)][0]);
      acc[t] = wmma_f16(a, bb, acc[t]);
    }
    __syncthreads();
  }

  int n  = lane & 15;
  int mh = (lane >> 4) << 3;
#pragma unroll
  for (int t = 0; t < 4; ++t) {
    int e = t * 16 + n;
    float bias_e = bias[h * DH + e];
#pragma unroll
    for (int r = 0; r < 8; ++r) {
      int m = r + mh;
      int s = s0 + wave * 16 + m;
      float val = acc[t][r] + bias_e;
      if constexpr (TRANS)
        out[((size_t)b * DH + e) * S + s] = (half_t)val;
      else
        out[(((size_t)b * HN + h) * S + s) * DH + e] = (half_t)val;
    }
  }
}

// ---------------------------------------------------------------------------
// Kernel 2: fused causal attention for one (b, h, 16-query tile).
//   K tiles TDM-staged (double-buffered) -> logits (f16 LDS) -> softmax ->
//   single-pass attn store (zeros for the masked tail) -> O = P * Vproj.
// ---------------------------------------------------------------------------
#define SPAD 2056   // 2048 + 8 halves padding
#define KPAD 72     // 64 + 8 halves: matches TDM pad_interval/amount

__global__ __launch_bounds__(128) void attn_kernel(
    const half_t* __restrict__ qh, const half_t* __restrict__ kh,
    const half_t* __restrict__ vT, float* __restrict__ attn_out,
    float* __restrict__ outh)
{
  __shared__ half_t sc[16][SPAD];
#ifdef HAVE_TDM
  __shared__ half_t ks[2][64][KPAD];
#endif
  __shared__ float  red[16][8];
  __shared__ float  rowmax[16];
  __shared__ float  rowsum[16];

  int tid = threadIdx.x, wave = tid >> 5, lane = tid & 31;
  int qt = blockIdx.x;  int s0 = qt * 16;
  int bh = blockIdx.y;  int b = bh / H, h = bh % H;
  int ktiles = qt + 1;                       // causal: keys 0..s0+15
  int keylen = ktiles * 16;
  int kpad   = ((ktiles + 1) & ~1) * 16;     // round up to 32-key chunks

  const half_t* qbase = qh + (((size_t)bh) * S + s0) * DH;
  const half_t* kbase = kh + ((size_t)bh) * S * DH;

  v16h aq0 = load_a_frag(qbase, DH, 0);
  v16h aq1 = load_a_frag(qbase, DH, 32);

  int n  = lane & 15;
  int mh = (lane >> 4) << 3;

  // ---- scores ------------------------------------------------------------
#ifdef HAVE_TDM
  int nchunks = (ktiles + 3) >> 2;           // 64 keys (4 tiles) per chunk
  if (wave == 0) {
    uint32_t rows0 = (uint32_t)min(64, keylen);
    tdm_load_k_tile((uint32_t)(uintptr_t)&ks[0][0][0], kbase, rows0);
  }
  for (int c = 0; c < nchunks; ++c) {
    if (wave == 0) __builtin_amdgcn_s_wait_tensorcnt(0);
    __syncthreads();                         // staged chunk c visible; buffer
                                             // (c+1)&1 no longer being read
    if (wave == 0 && c + 1 < nchunks) {
      int key0 = (c + 1) * 64;
      uint32_t rows = (uint32_t)min(64, keylen - key0);
      tdm_load_k_tile((uint32_t)(uintptr_t)&ks[(c + 1) & 1][0][0],
                      kbase + (size_t)key0 * DH, rows);
    }
    int kt = c * 4 + wave;
    if (kt < ktiles) {
      const half_t* kcol = &ks[c & 1][wave * 16 + n][0];
      v16h b0 = load_b_frag(kcol);
      v16h b1 = load_b_frag(kcol + 32);
      v8f c8 = (v8f){0,0,0,0,0,0,0,0};
      c8 = wmma_f16(aq0, b0, c8);
      c8 = wmma_f16(aq1, b1, c8);
#pragma unroll
      for (int r = 0; r < 8; ++r) {
        int m   = r + mh;
        int key = kt * 16 + n;
        float val = c8[r] * 0.125f;          // 1/sqrt(DH)
        if (key > s0 + m) val = -3.0e38f;    // -> -inf in f16 (masked)
        sc[m][kt * 16 + n] = (half_t)val;
      }
    }
  }
#else
  for (int kt = wave; kt < ktiles; kt += 4) {
    const half_t* kcol = kbase + (size_t)(kt * 16 + n) * DH;
    v16h b0 = load_b_frag(kcol);
    v16h b1 = load_b_frag(kcol + 32);
    v8f c8 = (v8f){0,0,0,0,0,0,0,0};
    c8 = wmma_f16(aq0, b0, c8);
    c8 = wmma_f16(aq1, b1, c8);
#pragma unroll
    for (int r = 0; r < 8; ++r) {
      int m   = r + mh;
      int key = kt * 16 + n;
      float val = c8[r] * 0.125f;
      if (key > s0 + m) val = -3.0e38f;
      sc[m][kt * 16 + n] = (half_t)val;
    }
  }
#endif
  if (ktiles & 1) {                          // zero-pad to a 32-key chunk
    for (int idx = tid; idx < 256; idx += 128)
      sc[idx >> 4][keylen + (idx & 15)] = (half_t)0.0f;
  }
  __syncthreads();

  // ---- softmax stats: 8 threads per row ----------------------------------
  int row = tid >> 3, t8 = tid & 7;
  float mx = -3.0e38f;
  for (int key = t8; key < keylen; key += 8)
    mx = fmaxf(mx, (float)sc[row][key]);
  red[row][t8] = mx;
  __syncthreads();
  if (tid < 16) {
    float mm = red[tid][0];
#pragma unroll
    for (int i = 1; i < 8; ++i) mm = fmaxf(mm, red[tid][i]);
    rowmax[tid] = mm;
  }
  __syncthreads();
  float mm = rowmax[row];
  float sum = 0.0f;
  for (int key = t8; key < keylen; key += 8) {
    float e = __expf((float)sc[row][key] - mm);   // -inf -> 0
    sc[row][key] = (half_t)e;                     // keep unnormalized p
    sum += e;
  }
  red[row][t8] = sum;
  __syncthreads();
  if (tid < 16) {
    float ss = 0.0f;
#pragma unroll
    for (int i = 0; i < 8; ++i) ss += red[tid][i];
    rowsum[tid] = ss;
  }
  __syncthreads();

  // ---- write the full attn row exactly once ------------------------------
  float inv = 1.0f / rowsum[row];
  float* arow = attn_out + (((size_t)b * S + s0 + row) * H + h) * S;
  for (int key = t8; key < S; key += 8) {
    float val = (key < keylen) ? (float)sc[row][key] * inv : 0.0f;
    arow[key] = val;
  }

  // ---- O = P * Vproj : wave w owns features [16w,16w+16) ------------------
  int f0 = wave * 16;
  v8f acc = (v8f){0,0,0,0,0,0,0,0};
  const half_t* vb = vT + ((size_t)b * DH + f0 + n) * S;  // transposed V
  for (int kc = 0; kc < kpad; kc += 32) {
    if (kc + 32 < kpad) __builtin_prefetch(vb + kc + 32, 0, 0);
    v16h a  = load_a_frag(&sc[0][0], SPAD, kc);
    v16h bb = load_b_frag(vb + kc);
    acc = wmma_f16(a, bb, acc);
  }
#pragma unroll
  for (int r = 0; r < 8; ++r) {
    int m = r + mh;
    float val = acc[r] / rowsum[m];          // normalize here
    outh[(((size_t)bh) * S + s0 + m) * DH + f0 + n] = val;
  }
}

// ---------------------------------------------------------------------------
// Kernel 3: mean over heads -> f16
// ---------------------------------------------------------------------------
__global__ void mean_kernel(const float* __restrict__ outh,
                            half_t* __restrict__ mean16)
{
  int idx = blockIdx.x * blockDim.x + threadIdx.x;    // B*S*DH
  if (idx >= B * S * DH) return;
  int b   = idx / (S * DH);
  int rem = idx % (S * DH);
  float acc = 0.0f;
#pragma unroll
  for (int h = 0; h < H; ++h)
    acc += outh[((size_t)b * H + h) * (S * DH) + rem];
  mean16[idx] = (half_t)(acc * (1.0f / H));
}

// ---------------------------------------------------------------------------
// Kernel 4: final projection out[b,s,d] = sum_e mean[b,s,e] * Wo[d,e]
// 16 s-rows x 256 d-cols per workgroup; each wave 16x64 (4 WMMA tiles), K=64.
// ---------------------------------------------------------------------------
__global__ __launch_bounds__(128) void outproj_kernel(
    const half_t* __restrict__ mean16, const float* __restrict__ Wo,
    float* __restrict__ out)
{
  int tid = threadIdx.x, wave = tid >> 5, lane = tid & 31;
  int s0 = blockIdx.x * 16;
  int b  = blockIdx.y;
  int dbase = blockIdx.z * 256 + wave * 64;

  const half_t* abase = mean16 + ((size_t)b * S + s0) * DH;
  v16h a0 = load_a_frag(abase, DH, 0);
  v16h a1 = load_a_frag(abase, DH, 32);

  int n = lane & 15, mh = (lane >> 4) << 3, kf = (lane >> 4) << 4;
#pragma unroll
  for (int t = 0; t < 4; ++t) {
    int dcol = dbase + t * 16 + n;
    const float* wrow = Wo + (size_t)dcol * DH;   // Wo[d, 0..63], contiguous
    v16h b0v, b1v;
#pragma unroll
    for (int r = 0; r < 8; ++r) {
      int k0 = kf + 2 * r;
      b0v[2 * r]     = (half_t)wrow[k0];
      b0v[2 * r + 1] = (half_t)wrow[k0 + 1];
      b1v[2 * r]     = (half_t)wrow[32 + k0];
      b1v[2 * r + 1] = (half_t)wrow[32 + k0 + 1];
    }
    v8f acc = (v8f){0,0,0,0,0,0,0,0};
    acc = wmma_f16(a0, b0v, acc);
    acc = wmma_f16(a1, b1v, acc);
#pragma unroll
    for (int r = 0; r < 8; ++r)
      out[((size_t)b * S + s0 + r + mh) * D + dcol] = acc[r];
  }
}

// ---------------------------------------------------------------------------
extern "C" void kernel_launch(void* const* d_in, const int* in_sizes, int n_in,
                              void* d_out, int out_size, void* d_ws, size_t ws_size,
                              hipStream_t stream) {
  const float* q  = (const float*)d_in[0];
  const float* k  = (const float*)d_in[1];
  const float* v  = (const float*)d_in[2];
  // d_in[3] = mask (causal; hardcoded in-kernel)
  const float* Wv = (const float*)d_in[4];
  const float* bv = (const float*)d_in[5];
  const float* Wq = (const float*)d_in[6];
  const float* bq = (const float*)d_in[7];
  const float* Wk = (const float*)d_in[8];
  const float* bk = (const float*)d_in[9];
  const float* Wo = (const float*)d_in[10];

  float* out  = (float*)d_out;                     // [B,S,D]
  float* attn = out + (size_t)B * S * D;           // [B,S,H,S]

  // workspace carve-out (~33 MB)
  half_t* qh16 = (half_t*)d_ws;                                 // [B,H,S,DH]
  half_t* kh16 = qh16 + (size_t)B * H * S * DH;                 // [B,H,S,DH]
  half_t* vT16 = kh16 + (size_t)B * H * S * DH;                 // [B,DH,S]
  float*  outh = (float*)(vT16 + (size_t)B * DH * S);           // [B,H,S,DH]
  half_t* mean16 = (half_t*)(outh + (size_t)B * H * S * DH);    // [B,S,DH]

  dim3 blk(128);
  proj_kernel<H, false><<<dim3(S / 64, B * H), blk, 0, stream>>>(q, Wq, bq, qh16);
  proj_kernel<H, false><<<dim3(S / 64, B * H), blk, 0, stream>>>(k, Wk, bk, kh16);
  proj_kernel<1, true ><<<dim3(S / 64, B),     blk, 0, stream>>>(v, Wv, bv, vT16);
  attn_kernel<<<dim3(S / 16, B * H), blk, 0, stream>>>(qh16, kh16, vT16, attn, outh);
  mean_kernel<<<dim3((B * S * DH + 255) / 256), dim3(256), 0, stream>>>(outh, mean16);
  outproj_kernel<<<dim3(S / 16, B, D / 256), blk, 0, stream>>>(mean16, Wo, out);
}
